// GNNStack_50775103373747
// MI455X (gfx1250) — compile-verified, matrix operand order checked by reading, not run on hardware
//
#include <hip/hip_runtime.h>
#include <hip/hip_bf16.h>
#include <math.h>

// ---------------- problem constants (match reference) ----------------
#define NNODES 100000
#define NEDGES 800000
#define NGRAPHS 512
#define IN_DIM 492
#define KP0 512          // IN_DIM padded to multiple of 32
#define HID 128
#define OUT_DIM 2
#define BN_EPS 1e-5f

typedef __attribute__((ext_vector_type(16))) __bf16 v16bf;
typedef __attribute__((ext_vector_type(8)))  __bf16 v8bf;
typedef __attribute__((ext_vector_type(8)))  float  v8f;

// ---------------------------------------------------------------------
// Cast f32 weight (K x HID) to bf16, zero-padded to Kp rows, and swizzle
// into WMMA B-fragment order:
//   out[(((kt*8 + nt)*32 + lane)*16 + e)] = W[k, nt*16 + (lane&15)]
//   k = kt*32 + (lane>>4)*8 + e + (e>=8 ? 8 : 0)
// so each lane's 16 fragment values are contiguous (32B) in memory/LDS.
__global__ void k_castswz(const float* __restrict__ w, __bf16* __restrict__ o,
                          int K, int Kp) {
  int i = blockIdx.x * blockDim.x + threadIdx.x;
  if (i >= Kp * HID) return;
  int e    = i & 15;
  int lane = (i >> 4) & 31;
  int nt   = (i >> 9) & 7;
  int kt   = i >> 12;
  int k    = kt * 32 + (lane >> 4) * 8 + e + ((e >= 8) ? 8 : 0);
  int col  = nt * 16 + (lane & 15);
  o[i] = (k < K) ? (__bf16)w[k * HID + col] : (__bf16)0.0f;
}

// ---------------------------------------------------------------------
// z[dst] += feat[src] over all edges (z pre-initialized to feat copy ->
// yields z = h + segment_sum(h[src], dst)).  One block per edge.
__global__ void k_edge_agg(const int* __restrict__ ei,
                           const float* __restrict__ feat,
                           float* __restrict__ z, int D) {
  int e = blockIdx.x;
  int src = ei[e];
  int dst = ei[NEDGES + e];
  const float* f = feat + (size_t)src * D;
  float* zz = z + (size_t)dst * D;
  for (int c = threadIdx.x; c < D; c += blockDim.x)
    unsafeAtomicAdd(&zz[c], f[c]);
}

// ---------------------------------------------------------------------
// out[n, 0:128] = act( Z[n, 0:K] @ W + bias ), W pre-swizzled bf16 (Kp x 128).
// act = ReLU then optional fused inference BN.
// ABF16: A activations stored bf16 (ld in elements); else f32.
// OBF16: store result as bf16 (no BN path used); else f32.
// Block: 256 threads = 8 waves; wave w computes rows [blk*128 + w*16, +16).
template <bool ABF16, bool OBF16>
__global__ void k_gemm(const void* __restrict__ Zv, int ldz, int K, int Kp,
                       const __bf16* __restrict__ Wswz,
                       const float* __restrict__ bias,
                       const float* __restrict__ gamma,
                       const float* __restrict__ beta,
                       const float* __restrict__ rm,
                       const float* __restrict__ rv,
                       int do_bn, float* __restrict__ outf,
                       __bf16* __restrict__ outb, int nrows) {
  extern __shared__ __bf16 lw[];      // Kp*128 bf16, fragment order
  const int tid = threadIdx.x;

  // cooperative 16B-chunk copy of the swizzled weight panel into LDS
  {
    const int nchunk = (Kp * HID) / 8;             // 8 bf16 per 16B
    const uint4* gsrc = (const uint4*)Wswz;
    uint4* ldst = (uint4*)lw;
    for (int i = tid; i < nchunk; i += blockDim.x) ldst[i] = gsrc[i];
  }
  __syncthreads();

  const int lane = tid & 31;
  const int wave = tid >> 5;
  const int half = lane >> 4;
  const int l16  = lane & 15;
  const int m0   = blockIdx.x * 128 + wave * 16;

  int mrow = m0 + l16;
  if (mrow >= nrows) mrow = nrows - 1;             // clamp loads; stores guarded
  const float*  zrowf = ABF16 ? nullptr : (const float*)Zv + (size_t)mrow * ldz;
  const __bf16* zrowb = ABF16 ? (const __bf16*)Zv + (size_t)mrow * ldz : nullptr;

  v8f acc[8];
#pragma unroll
  for (int t = 0; t < 8; ++t)
#pragma unroll
    for (int r = 0; r < 8; ++r) acc[t][r] = 0.0f;

  const int nfull = K >> 5;            // branch-free full K-tiles
  const __bf16* frag = lw + (lane << 4);

#pragma unroll 1
  for (int kt = 0; kt < nfull; ++kt) {
    const int kbase = (kt << 5) + half * 8;
    // ---- A fragment (16x32 bf16): element e -> k = half*8 + e (+8 if e>=8)
    v16bf a;
    if (ABF16) {
      v8bf lo = *(const v8bf*)(zrowb + kbase);
      v8bf hi = *(const v8bf*)(zrowb + kbase + 16);
      a = __builtin_shufflevector(lo, hi, 0, 1, 2, 3, 4, 5, 6, 7,
                                  8, 9, 10, 11, 12, 13, 14, 15);
    } else {
      const float4* p = (const float4*)(zrowf + kbase);        // k .. k+7
      const float4* q = (const float4*)(zrowf + kbase + 16);   // k+16 .. k+23
      float4 f0 = p[0], f1 = p[1], f2 = q[0], f3 = q[1];
      a[0] = (__bf16)f0.x;  a[1] = (__bf16)f0.y;  a[2]  = (__bf16)f0.z;  a[3]  = (__bf16)f0.w;
      a[4] = (__bf16)f1.x;  a[5] = (__bf16)f1.y;  a[6]  = (__bf16)f1.z;  a[7]  = (__bf16)f1.w;
      a[8] = (__bf16)f2.x;  a[9] = (__bf16)f2.y;  a[10] = (__bf16)f2.z;  a[11] = (__bf16)f2.w;
      a[12] = (__bf16)f3.x; a[13] = (__bf16)f3.y; a[14] = (__bf16)f3.z;  a[15] = (__bf16)f3.w;
    }

    // ---- issue all 16 ds_load_b128 first, then 8 back-to-back WMMAs
    v16bf bfr[8];
#pragma unroll
    for (int nt = 0; nt < 8; ++nt) {
      v8bf blo = *(const v8bf*)(frag + nt * 512);
      v8bf bhi = *(const v8bf*)(frag + nt * 512 + 8);
      bfr[nt] = __builtin_shufflevector(blo, bhi, 0, 1, 2, 3, 4, 5, 6, 7,
                                        8, 9, 10, 11, 12, 13, 14, 15);
    }
#pragma unroll
    for (int nt = 0; nt < 8; ++nt)
      acc[nt] = __builtin_amdgcn_wmma_f32_16x16x32_bf16(
          false, a, false, bfr[nt], (short)0, acc[nt], false, false);
    frag += 4096;
  }

  // ---- single guarded tail tile (only when K not multiple of 32) ----
  if ((nfull << 5) < Kp) {
    const int kbase = (nfull << 5) + half * 8;
    v16bf a;
#pragma unroll
    for (int e = 0; e < 16; ++e) {
      int k = kbase + e + ((e >= 8) ? 8 : 0);
      if (ABF16) a[e] = (k < K) ? zrowb[k] : (__bf16)0.0f;
      else       a[e] = (k < K) ? (__bf16)zrowf[k] : (__bf16)0.0f;
    }
    v16bf bfr[8];
#pragma unroll
    for (int nt = 0; nt < 8; ++nt) {
      v8bf blo = *(const v8bf*)(frag + nt * 512);
      v8bf bhi = *(const v8bf*)(frag + nt * 512 + 8);
      bfr[nt] = __builtin_shufflevector(blo, bhi, 0, 1, 2, 3, 4, 5, 6, 7,
                                        8, 9, 10, 11, 12, 13, 14, 15);
    }
#pragma unroll
    for (int nt = 0; nt < 8; ++nt)
      acc[nt] = __builtin_amdgcn_wmma_f32_16x16x32_bf16(
          false, a, false, bfr[nt], (short)0, acc[nt], false, false);
  }

  // Epilogue: C/D layout -> VGPR r holds row m0 + half*8 + r, col = nt*16 + l16
  const int rbase = m0 + half * 8;
#pragma unroll
  for (int nt = 0; nt < 8; ++nt) {
    int col = nt * 16 + l16;
    float bv = bias[col];
    float sc = 1.0f, sh = 0.0f;
    if (!OBF16 && do_bn) {
      float inv = rsqrtf(rv[col] + BN_EPS);
      sc = gamma[col] * inv;
      sh = beta[col] - rm[col] * sc;
    }
#pragma unroll
    for (int r = 0; r < 8; ++r) {
      int row = rbase + r;
      if (row < nrows) {
        float v = fmaxf(acc[nt][r] + bv, 0.0f);
        if (OBF16) {
          outb[(size_t)row * HID + col] = (__bf16)v;
        } else {
          if (do_bn) v = v * sc + sh;
          outf[(size_t)row * HID + col] = v;
        }
      }
    }
  }
}

// ---------------------------------------------------------------------
// Graph mean-pool accumulation (sums, counts). One block (128 thr) per node.
__global__ void k_pool(const float* __restrict__ h, const int* __restrict__ batch,
                       float* __restrict__ sums, float* __restrict__ cnt) {
  int n = blockIdx.x;
  int g = batch[n];
  unsafeAtomicAdd(&sums[(size_t)g * HID + threadIdx.x],
                  h[(size_t)n * HID + threadIdx.x]);
  if (threadIdx.x == 0) unsafeAtomicAdd(&cnt[g], 1.0f);
}

// ---------------------------------------------------------------------
// Per graph: emb = sums/max(cnt,1); out = log_softmax((emb@Wp1+bp1)@Wp2+bp2)
__global__ void k_postmp(const float* __restrict__ sums, const float* __restrict__ cnt,
                         const float* __restrict__ Wp1, const float* __restrict__ bp1,
                         const float* __restrict__ Wp2, const float* __restrict__ bp2,
                         float* __restrict__ out_emb, float* __restrict__ out_log) {
  __shared__ float semb[HID];
  __shared__ float sh1[HID];
  __shared__ float sz[OUT_DIM];
  int g = blockIdx.x;
  int c = threadIdx.x;
  float cn = fmaxf(cnt[g], 1.0f);
  float e = sums[(size_t)g * HID + c] / cn;
  semb[c] = e;
  out_emb[(size_t)g * HID + c] = e;
  __syncthreads();
  float acc = bp1[c];
  for (int k = 0; k < HID; ++k) acc += semb[k] * Wp1[k * HID + c];
  sh1[c] = acc;
  __syncthreads();
  if (c < OUT_DIM) {
    float z = bp2[c];
    for (int k = 0; k < HID; ++k) z += sh1[k] * Wp2[k * OUT_DIM + c];
    sz[c] = z;
  }
  __syncthreads();
  if (c == 0) {
    float mx = fmaxf(sz[0], sz[1]);
    float lse = mx + logf(expf(sz[0] - mx) + expf(sz[1] - mx));
    out_log[(size_t)g * OUT_DIM + 0] = sz[0] - lse;
    out_log[(size_t)g * OUT_DIM + 1] = sz[1] - lse;
  }
}

// ---------------------------------------------------------------------
extern "C" void kernel_launch(void* const* d_in, const int* in_sizes, int n_in,
                              void* d_out, int out_size, void* d_ws, size_t ws_size,
                              hipStream_t stream) {
  (void)in_sizes; (void)n_in; (void)out_size; (void)ws_size;

  const float* x     = (const float*)d_in[0];
  const int*   ei    = (const int*)d_in[1];
  const int*   batch = (const int*)d_in[2];
  const float* Wa[3] = {(const float*)d_in[3], (const float*)d_in[7],  (const float*)d_in[11]};
  const float* ba[3] = {(const float*)d_in[4], (const float*)d_in[8],  (const float*)d_in[12]};
  const float* Wb[3] = {(const float*)d_in[5], (const float*)d_in[9],  (const float*)d_in[13]};
  const float* bb[3] = {(const float*)d_in[6], (const float*)d_in[10], (const float*)d_in[14]};
  const float* gam[2] = {(const float*)d_in[15], (const float*)d_in[19]};
  const float* bet[2] = {(const float*)d_in[16], (const float*)d_in[20]};
  const float* rm [2] = {(const float*)d_in[17], (const float*)d_in[21]};
  const float* rv [2] = {(const float*)d_in[18], (const float*)d_in[22]};
  const float* Wp1 = (const float*)d_in[23];
  const float* bp1 = (const float*)d_in[24];
  const float* Wp2 = (const float*)d_in[25];
  const float* bp2 = (const float*)d_in[26];
  float* out = (float*)d_out;

  // ---- workspace carving (256B aligned) ----
  char* ws = (char*)d_ws;
  size_t off = 0;
  auto carve = [&](size_t bytes) -> void* {
    off = (off + 255) & ~(size_t)255;
    void* p = ws + off;
    off += bytes;
    return p;
  };
  float*  z0   = (float*)carve((size_t)NNODES * IN_DIM * 4);  // layer0 aggregate
  __bf16* tbf  = (__bf16*)carve((size_t)NNODES * HID * 2);    // hidden t (bf16)
  float*  bufB = (float*)carve((size_t)NNODES * HID * 4);     // layer output h
  float*  bufC = (float*)carve((size_t)NNODES * HID * 4);     // z for layers 1/2
  __bf16* wa0p = (__bf16*)carve((size_t)KP0 * HID * 2);
  __bf16* wb0p = (__bf16*)carve((size_t)HID * HID * 2);
  __bf16* wa1p = (__bf16*)carve((size_t)HID * HID * 2);
  __bf16* wb1p = (__bf16*)carve((size_t)HID * HID * 2);
  __bf16* wa2p = (__bf16*)carve((size_t)HID * HID * 2);
  __bf16* wb2p = (__bf16*)carve((size_t)HID * HID * 2);
  float*  sums = (float*)carve((size_t)(NGRAPHS * HID + NGRAPHS) * 4);
  float*  cnt  = sums + (size_t)NGRAPHS * HID;

  const int GEMM_GRID = (NNODES + 127) / 128;
  const int CAST_BLK = 256;

  // ---- weight conversion to padded, fragment-swizzled bf16 ----
  k_castswz<<<(KP0 * HID + CAST_BLK - 1) / CAST_BLK, CAST_BLK, 0, stream>>>(Wa[0], wa0p, IN_DIM, KP0);
  k_castswz<<<(HID * HID + CAST_BLK - 1) / CAST_BLK, CAST_BLK, 0, stream>>>(Wb[0], wb0p, HID, HID);
  k_castswz<<<(HID * HID + CAST_BLK - 1) / CAST_BLK, CAST_BLK, 0, stream>>>(Wa[1], wa1p, HID, HID);
  k_castswz<<<(HID * HID + CAST_BLK - 1) / CAST_BLK, CAST_BLK, 0, stream>>>(Wb[1], wb1p, HID, HID);
  k_castswz<<<(HID * HID + CAST_BLK - 1) / CAST_BLK, CAST_BLK, 0, stream>>>(Wa[2], wa2p, HID, HID);
  k_castswz<<<(HID * HID + CAST_BLK - 1) / CAST_BLK, CAST_BLK, 0, stream>>>(Wb[2], wb2p, HID, HID);

  // ---- layer 0 ----
  hipMemcpyAsync(z0, x, (size_t)NNODES * IN_DIM * 4, hipMemcpyDeviceToDevice, stream);
  k_edge_agg<<<NEDGES, 128, 0, stream>>>(ei, x, z0, IN_DIM);
  k_gemm<false, true><<<GEMM_GRID, 256, KP0 * HID * 2, stream>>>(z0, IN_DIM, IN_DIM, KP0,
      wa0p, ba[0], nullptr, nullptr, nullptr, nullptr, 0, nullptr, tbf, NNODES);
  k_gemm<true, false><<<GEMM_GRID, 256, HID * HID * 2, stream>>>(tbf, HID, HID, HID,
      wb0p, bb[0], gam[0], bet[0], rm[0], rv[0], 1, bufB, nullptr, NNODES);

  // ---- layer 1 ----
  hipMemcpyAsync(bufC, bufB, (size_t)NNODES * HID * 4, hipMemcpyDeviceToDevice, stream);
  k_edge_agg<<<NEDGES, 128, 0, stream>>>(ei, bufB, bufC, HID);
  k_gemm<false, true><<<GEMM_GRID, 256, HID * HID * 2, stream>>>(bufC, HID, HID, HID,
      wa1p, ba[1], nullptr, nullptr, nullptr, nullptr, 0, nullptr, tbf, NNODES);
  k_gemm<true, false><<<GEMM_GRID, 256, HID * HID * 2, stream>>>(tbf, HID, HID, HID,
      wb1p, bb[1], gam[1], bet[1], rm[1], rv[1], 1, bufB, nullptr, NNODES);

  // ---- layer 2 ----
  hipMemcpyAsync(bufC, bufB, (size_t)NNODES * HID * 4, hipMemcpyDeviceToDevice, stream);
  k_edge_agg<<<NEDGES, 128, 0, stream>>>(ei, bufB, bufC, HID);
  k_gemm<false, true><<<GEMM_GRID, 256, HID * HID * 2, stream>>>(bufC, HID, HID, HID,
      wa2p, ba[2], nullptr, nullptr, nullptr, nullptr, 0, nullptr, tbf, NNODES);
  k_gemm<true, false><<<GEMM_GRID, 256, HID * HID * 2, stream>>>(tbf, HID, HID, HID,
      wb2p, bb[2], nullptr, nullptr, nullptr, nullptr, 0, bufB, nullptr, NNODES);

  // ---- global mean pool + post MLP + log_softmax ----
  hipMemsetAsync(sums, 0, (size_t)(NGRAPHS * HID + NGRAPHS) * 4, stream);
  k_pool<<<NNODES, 128, 0, stream>>>(bufB, batch, sums, cnt);
  k_postmp<<<NGRAPHS, 128, 0, stream>>>(sums, cnt, Wp1, bp1, Wp2, bp2,
                                        out, out + (size_t)NGRAPHS * HID);
}
